// STPWrapper_27273042329926
// MI455X (gfx1250) — matrix operand order repaired and checked by reading, not compiled, not run on hardware
//
#include <hip/hip_runtime.h>

// ---------------- problem constants ----------------
#define NN      2048
#define SEQ     500
#define BB      16
#define NBLK    128            // NN/16 tiles, one wave each (co-resident)

#define DT      1e-4f
#define U_CONST 0.3f
#define INV_TAU   125.0f       // 1/0.008
#define INV_TAU_F (1.0f/1.5f)
#define INV_TAU_D (1.0f/0.3f)
#define ALPHA   1.5f
#define I_B     8.0f
#define J_EI    1.1f
#define J_IE    2.2f

// ---------------- CDNA5 WMMA types ----------------
typedef __bf16        v16bf __attribute__((ext_vector_type(16)));
typedef float         v8f   __attribute__((ext_vector_type(8)));
typedef unsigned int  u32x4 __attribute__((ext_vector_type(4)));

union FragBF {
    v16bf bf;
    u32x4 q[2];
};

__device__ __forceinline__ float splus(float h) {
    float z = h * (1.0f / ALPHA);
    float sp = (z > 30.0f) ? z : log1pf(expf(z));
    return ALPHA * sp;
}

__device__ __forceinline__ unsigned short f2bf(float f) {
    unsigned int v = __float_as_uint(f);
    v += 0x7FFFu + ((v >> 16) & 1u);
    return (unsigned short)(v >> 16);
}

// agent-scope grid barrier (128 one-wave workgroups, all resident)
__device__ __forceinline__ void grid_barrier(unsigned* cnt, unsigned* gen) {
    if (threadIdx.x == 0) {
        __threadfence();   // make vbuf/sumR/hI stores visible device-wide
        unsigned g = __hip_atomic_load(gen, __ATOMIC_RELAXED, __HIP_MEMORY_SCOPE_AGENT);
        unsigned a = __hip_atomic_fetch_add(cnt, 1u, __ATOMIC_ACQ_REL, __HIP_MEMORY_SCOPE_AGENT);
        if (a == NBLK - 1u) {
            __hip_atomic_store(cnt, 0u, __ATOMIC_RELAXED, __HIP_MEMORY_SCOPE_AGENT);
            __hip_atomic_store(gen, g + 1u, __ATOMIC_RELEASE, __HIP_MEMORY_SCOPE_AGENT);
        } else {
            while (__hip_atomic_load(gen, __ATOMIC_ACQUIRE, __HIP_MEMORY_SCOPE_AGENT) == g)
                __builtin_amdgcn_s_sleep(1);
        }
    }
    __syncthreads();
}

// ---------------- init: convert J (f32) -> Jt (bf16, [n][k], no transpose) ----------------
__global__ __launch_bounds__(256)
void convert_J(const float* __restrict__ J, unsigned short* __restrict__ Jt,
               unsigned* __restrict__ bar) {
    __shared__ float tile[32][33];
    const int tx = threadIdx.x, ty = threadIdx.y;
    const int kb = blockIdx.x * 32, nb = blockIdx.y * 32;
    #pragma unroll
    for (int j = 0; j < 32; j += 8)
        tile[ty + j][tx] = J[(size_t)(nb + ty + j) * NN + (kb + tx)];
    __syncthreads();
    #pragma unroll
    for (int j = 0; j < 32; j += 8)
        Jt[(size_t)(nb + ty + j) * NN + (kb + tx)] = f2bf(tile[ty + j][tx]);
    if (blockIdx.x == 0 && blockIdx.y == 0 && tx == 0 && ty == 0) {
        bar[0] = 0u;   // arrival counter
        bar[1] = 0u;   // generation
    }
}

// ---------------- persistent kernel: all 500 steps, 1 launch ----------------
__global__ __launch_bounds__(32)
void stp_persistent(const float* __restrict__ inp, const float* __restrict__ W_in,
                    const float* __restrict__ W_out,
                    const unsigned short* __restrict__ Jt,
                    unsigned short* __restrict__ vbuf,   // 2 x BB*NN bf16 (ping-pong A)
                    float* __restrict__ sumRbuf,         // 3 x BB f32 (rotating)
                    unsigned* __restrict__ bar,
                    float* __restrict__ hs, float* __restrict__ us,
                    float* __restrict__ xs, float* __restrict__ hIs,
                    float* __restrict__ outs) {
    const int lane = threadIdx.x;
    const int blk  = blockIdx.x;
    const int half = lane >> 4;          // 0: K chunks 0..7/16..23, 1: 8..15/24..31
    const int mrow = lane & 15;
    const int n    = blk * 16 + mrow;    // this lane's neuron column
    const float win  = W_in[n];
    const float wout = W_out[n];
    const size_t strideT = (size_t)BB * NN;

    // ---- register-resident state: batches m = r + half*8 at neuron n ----
    float h[8], u[8], x[8], hI[8];
    #pragma unroll
    for (int r = 0; r < 8; ++r) { h[r] = 0.0f; u[r] = U_CONST; x[r] = 1.0f; hI[r] = 0.0f; }

    // ---- init phase: A-matrix for t=0, sumR rotation, zero outs ----
    const float R0 = splus(0.0f);                    // 1.5*ln2
    const unsigned short v0 = f2bf(U_CONST * R0);    // u0*x0*R0
    #pragma unroll
    for (int r = 0; r < 8; ++r)
        vbuf[(size_t)(r + half * 8) * NN + n] = v0;
    if (blk == 0 && lane < BB) {
        sumRbuf[0 * BB + lane] = (float)NN * R0;     // read at t=0
        sumRbuf[1 * BB + lane] = 0.0f;               // accum target at t=0
        sumRbuf[2 * BB + lane] = 0.0f;               // zero target at t=0
    }
    for (int i = blk * 32 + lane; i < SEQ * BB; i += NBLK * 32) outs[i] = 0.0f;

    grid_barrier(bar, bar + 1);

    for (int t = 0; t < SEQ; ++t) {
        const unsigned short* vb = vbuf + (size_t)(t & 1) * strideT;
        unsigned short* vnext    = vbuf + (size_t)((t + 1) & 1) * strideT;

        // ---- GEMM: syn(16x16 tile) = v(16xK) x Jt(Kx16), bf16 WMMA, f32 acc ----
        v8f c = {0.f, 0.f, 0.f, 0.f, 0.f, 0.f, 0.f, 0.f};
        const unsigned short* aBase = vb + (size_t)mrow * NN + half * 8;
        for (int kk = 0; kk < NN; kk += 32) {
            FragBF a, b;
            const u32x4* ap = (const u32x4*)(aBase + kk);
            a.q[0] = ap[0];                 // K = kk+c .. kk+c+7
            a.q[1] = ap[2];                 // K = kk+c+16 .. kk+c+23
            const u32x4* bp = (const u32x4*)(Jt + (size_t)n * NN + kk + half * 8);
            b.q[0] = bp[0];
            b.q[1] = bp[2];
            c = __builtin_amdgcn_wmma_f32_16x16x32_bf16(
                    false, a.bf, false, b.bf, (short)0, c, false, false);
        }

        // ---- elementwise Euler update, trajectory stores, next-step A ----
        float contrib[8], rsum[8];
        #pragma unroll
        for (int r = 0; r < 8; ++r) {
            const int m = r + half * 8;
            float R   = splus(h[r]);
            float R_I = splus(hI[r]);
            float Ie  = inp[t * BB + m] * win;
            float h2 = h[r] + DT * ((-h[r] + c[r] - J_EI * R_I + I_B + Ie) * INV_TAU);
            float u2 = u[r] + DT * ((U_CONST - u[r]) * INV_TAU_F + U_CONST * (1.0f - u[r]) * R);
            float x2 = x[r] + DT * ((1.0f - x[r]) * INV_TAU_D - u[r] * x[r] * R);
            size_t o = (size_t)t * strideT + (size_t)m * NN + n;
            __builtin_nontemporal_store(h2, hs + o);   // streaming: keep Jt in L2
            __builtin_nontemporal_store(u2, us + o);
            __builtin_nontemporal_store(x2, xs + o);
            float R2 = splus(h2);
            vnext[(size_t)m * NN + n] = f2bf(u2 * x2 * R2);
            contrib[r] = R2 * wout;
            rsum[r]    = R2;
            h[r] = h2; u[r] = u2; x[r] = x2;
        }

        // ---- h_I update, replicated in every block (identical arithmetic) ----
        #pragma unroll
        for (int r = 0; r < 8; ++r) {
            float sR = sumRbuf[(t % 3) * BB + (r + half * 8)];
            hI[r] = hI[r] + DT * ((-hI[r] + J_IE * sR) * INV_TAU);
        }
        if (blk == 0 && mrow == 0) {           // lanes 0 and 16 write trajectory
            #pragma unroll
            for (int r = 0; r < 8; ++r)
                hIs[t * BB + half * 8 + r] = hI[r];
        }

        // ---- reductions across the 16-lane halves (per-batch sums) ----
        #pragma unroll
        for (int r = 0; r < 8; ++r) {
            float s = contrib[r], q = rsum[r];
            s += __shfl_xor(s, 1, 32);  q += __shfl_xor(q, 1, 32);
            s += __shfl_xor(s, 2, 32);  q += __shfl_xor(q, 2, 32);
            s += __shfl_xor(s, 4, 32);  q += __shfl_xor(q, 4, 32);
            s += __shfl_xor(s, 8, 32);  q += __shfl_xor(q, 8, 32);
            if (mrow == 0) {
                atomicAdd(&outs[t * BB + half * 8 + r], s);
                atomicAdd(&sumRbuf[((t + 1) % 3) * BB + half * 8 + r], q);
            }
        }
        // zero the buffer that becomes the accumulate target next step
        if (blk == 0 && lane < BB)
            sumRbuf[((t + 2) % 3) * BB + lane] = 0.0f;

        grid_barrier(bar, bar + 1);
    }
}

// ---------------- host orchestration: 2 launches total ----------------
extern "C" void kernel_launch(void* const* d_in, const int* in_sizes, int n_in,
                              void* d_out, int out_size, void* d_ws, size_t ws_size,
                              hipStream_t stream) {
    const float* inp   = (const float*)d_in[0];   // (SEQ, B, 1)
    const float* W_in  = (const float*)d_in[1];   // (N, 1)
    const float* J     = (const float*)d_in[2];   // (N, N)
    const float* W_out = (const float*)d_in[3];   // (1, N)

    float* out  = (float*)d_out;
    float* hs   = out;
    float* us   = hs + (size_t)SEQ * BB * NN;
    float* xs   = us + (size_t)SEQ * BB * NN;
    float* hIs  = xs + (size_t)SEQ * BB * NN;
    float* outs = hIs + (size_t)SEQ * BB;

    unsigned short* Jt   = (unsigned short*)d_ws;            // N*N bf16 = 8 MB
    unsigned short* vbuf = Jt + (size_t)NN * NN;             // 2 * B*N bf16
    float* sumRbuf = (float*)(vbuf + 2 * (size_t)BB * NN);   // 3 * B f32
    unsigned* bar  = (unsigned*)(sumRbuf + 3 * BB);          // counter + generation

    convert_J<<<dim3(NN / 32, NN / 32), dim3(32, 8), 0, stream>>>(J, Jt, bar);
    stp_persistent<<<NBLK, 32, 0, stream>>>(inp, W_in, W_out, Jt, vbuf, sumRbuf,
                                            bar, hs, us, xs, hIs, outs);
}
